// ViT_8641474200324
// MI455X (gfx1250) — matrix operand order repaired and checked by reading, not compiled, not run on hardware
//
#include <hip/hip_runtime.h>
#include <hip/hip_bf16.h>
#include <math.h>

// ---------------------------------------------------------------------------
// ViT-Base forward for MI455X (gfx1250, wave32, WMMA f32_16x16x32_f16)
// ---------------------------------------------------------------------------
#define Bb   32
#define Cc   3
#define IMG  224
#define Pp   16
#define Dd   768
#define Ll   12
#define Hh   12
#define DH   64
#define EXPD 3072
#define NCLS 1000
#define NP   196          // patches per image
#define Nt   197          // tokens
#define Mtok (Bb*Nt)      // 6304
#define Mpat (Bb*NP)      // 6272
#define BH   (Bb*Hh)      // 384
#define NPAD 208          // 13*16 token pad (rows)
#define KPAD 224          // 7*32  key pad (K dim for P@V)

typedef __attribute__((ext_vector_type(16))) _Float16 v16h;
typedef __attribute__((ext_vector_type(8)))  float    v8f;

// ------------------------- WMMA fragment loaders ---------------------------
// A (16x32 f16), row-major source with row stride ld (elements).
// lane l: m = l%16, g = l/16 ; element s -> K = g*16 + s
__device__ __forceinline__ v16h load_a_frag(const _Float16* __restrict__ A, int ld) {
  int lane = threadIdx.x & 31;
  int m = lane & 15, g = lane >> 4;
  const _Float16* row = A + m * ld;
  v16h a;
#pragma unroll
  for (int v = 0; v < 4; ++v) {
    a[2*v]     = row[g*16 + 2*v];
    a[2*v + 1] = row[g*16 + 2*v + 1];
    a[8 + 2*v]     = row[g*16 + 8 + 2*v];
    a[8 + 2*v + 1] = row[g*16 + 8 + 2*v + 1];
  }
  return a;
}

// B (32x16 f16) where source holds B^T: W[n][k], row-major, stride ld.
// lane l: n = l%16, g = l/16 (g=0 -> K 0..15, g=1 -> K 16..31), VGPR v -> K = g*16+2v,+1
__device__ __forceinline__ v16h load_b_frag_wt(const _Float16* __restrict__ W, int ld) {
  int lane = threadIdx.x & 31;
  int n = lane & 15, g = lane >> 4;
  const _Float16* row = W + n * ld + g * 16;
  v16h b;
#pragma unroll
  for (int v = 0; v < 8; ++v) {
    b[2*v]     = row[2*v];
    b[2*v + 1] = row[2*v + 1];
  }
  return b;
}

// B (32x16 f16) where source is row-major [K][N] with row stride ld.
__device__ __forceinline__ v16h load_b_frag_kn(const _Float16* __restrict__ Bm, int ld) {
  int lane = threadIdx.x & 31;
  int n = lane & 15, g = lane >> 4;
  v16h b;
#pragma unroll
  for (int v = 0; v < 8; ++v) {
    b[2*v]     = Bm[(g*16 + 2*v)     * ld + n];
    b[2*v + 1] = Bm[(g*16 + 2*v + 1) * ld + n];
  }
  return b;
}

#define WMMA_F16(a, b, c) \
  __builtin_amdgcn_wmma_f32_16x16x32_f16(false, (a), false, (b), (short)0, (c), false, false)

__device__ __forceinline__ float gelu_exact(float x) {
  return 0.5f * x * (1.0f + erff(x * 0.70710678118654752f));
}

// ------------------------- generic WMMA GEMM -------------------------------
// C[M,N] = act(A[M,K] @ W[N,K]^T + bias) + res
// block = 256 thr (8 waves), tile 128x128, wave tile 32x64, K-step 32.
// Software pipeline: LDS store(k) -> barrier -> issue global loads(k+32)
// -> 8 WMMA from LDS -> barrier.  K must be a multiple of 32.
#define LDK 36   // f16 LDS row stride (bank-conflict padding)

__global__ __launch_bounds__(256)
void gemm_bias_kernel(const float* __restrict__ A, const float* __restrict__ W,
                      const float* __restrict__ bias, const float* __restrict__ res,
                      float* __restrict__ C, int M, int N, int K, int act)
{
  __shared__ _Float16 As[128 * LDK];
  __shared__ _Float16 Bs[128 * LDK];
  const int tid  = threadIdx.x;
  const int wave = tid >> 5;
  const int mg   = wave >> 1;        // 0..3 : 32-row group
  const int ng   = wave & 1;         // 0..1 : 64-col group
  const int m0 = blockIdx.x * 128;
  const int n0 = blockIdx.y * 128;

  const int tr = tid >> 3;           // 0..31 row within a 32-row pass
  const int tc = (tid & 7) * 4;      // f32 column 0..28

  v8f acc[2][4] = {};
  float4 ra[4], rb[4];

  // branch-free staging load: clamp row, select-zero out-of-range
  auto gload = [&](int k0) {
#pragma unroll
    for (int p = 0; p < 4; ++p) {
      int r = p * 32 + tr;
      int gm = m0 + r;
      int gmc = gm < M ? gm : M - 1;
      float4 v = *(const float4*)(A + (size_t)gmc * K + k0 + tc);
      if (gm >= M) v = make_float4(0.f, 0.f, 0.f, 0.f);
      ra[p] = v;
      int gn = n0 + r;
      int gnc = gn < N ? gn : N - 1;
      float4 w = *(const float4*)(W + (size_t)gnc * K + k0 + tc);
      if (gn >= N) w = make_float4(0.f, 0.f, 0.f, 0.f);
      rb[p] = w;
    }
  };
  auto lstore = [&]() {
#pragma unroll
    for (int p = 0; p < 4; ++p) {
      int r = p * 32 + tr;
      _Float16* da = As + r * LDK + tc;
      da[0] = (_Float16)ra[p].x; da[1] = (_Float16)ra[p].y;
      da[2] = (_Float16)ra[p].z; da[3] = (_Float16)ra[p].w;
      _Float16* db = Bs + r * LDK + tc;
      db[0] = (_Float16)rb[p].x; db[1] = (_Float16)rb[p].y;
      db[2] = (_Float16)rb[p].z; db[3] = (_Float16)rb[p].w;
    }
  };

  gload(0);                          // prologue
  for (int k0 = 0; k0 < K; k0 += 32) {
    lstore();                        // regs(k0) -> LDS
    __syncthreads();
    if (k0 + 32 < K) gload(k0 + 32); // loads for next step fly during WMMA
    v16h a0 = load_a_frag(As + (mg * 32)      * LDK, LDK);
    v16h a1 = load_a_frag(As + (mg * 32 + 16) * LDK, LDK);
#pragma unroll
    for (int nt = 0; nt < 4; ++nt) {
      v16h b = load_b_frag_wt(Bs + (ng * 64 + nt * 16) * LDK, LDK);
      acc[0][nt] = WMMA_F16(a0, b, acc[0][nt]);
      acc[1][nt] = WMMA_F16(a1, b, acc[1][nt]);
    }
    __syncthreads();
  }

  // store: lane l holds col n = l%16, rows (l/16)*8 + v   (16x16 f32 C layout)
  const int lane = tid & 31;
  const int nl = lane & 15, g = lane >> 4;
#pragma unroll
  for (int mt = 0; mt < 2; ++mt) {
#pragma unroll
    for (int nt = 0; nt < 4; ++nt) {
      int gn = n0 + ng * 64 + nt * 16 + nl;
      if (gn >= N) continue;
      float bv = bias ? bias[gn] : 0.f;
#pragma unroll
      for (int v = 0; v < 8; ++v) {
        int gm = m0 + mg * 32 + mt * 16 + g * 8 + v;
        if (gm >= M) continue;
        float x = acc[mt][nt][v] + bv;
        if (act == 1) x = gelu_exact(x);
        else if (act == 2) x = tanhf(x);
        if (res) x += res[(size_t)gm * N + gn];
        C[(size_t)gm * N + gn] = x;
      }
    }
  }
}

// ------------------------- elementwise / prep kernels ----------------------
__global__ void im2col_kernel(const float* __restrict__ img, float* __restrict__ X) {
  int idx = blockIdx.x * 256 + threadIdx.x;
  if (idx >= Mpat * Dd) return;
  int j = idx % Dd, t = idx / Dd;
  int b = t / NP, patch = t % NP;
  int gy = patch / 14, gx = patch % 14;
  int c = j % Cc, pp = j / Cc;
  int py = pp / Pp, px = pp % Pp;
  X[idx] = img[(((size_t)b * Cc + c) * IMG + gy * Pp + py) * IMG + gx * Pp + px];
}

__global__ void assemble_kernel(const float* __restrict__ Pm, const float* __restrict__ cls,
                                const float* __restrict__ pos, float* __restrict__ x) {
  int idx = blockIdx.x * 256 + threadIdx.x;
  if (idx >= Mtok * Dd) return;
  int d = idx % Dd, t = idx / Dd;
  int b = t / Nt, n = t % Nt;
  float v = pos[n * Dd + d];
  if (n == 0) v += cls[d];
  else        v += Pm[((size_t)b * NP + (n - 1)) * Dd + d];
  x[idx] = v;
}

__global__ __launch_bounds__(256)
void layernorm_kernel(const float* __restrict__ x, const float* __restrict__ w,
                      const float* __restrict__ b, float* __restrict__ out) {
  __shared__ float ssum[256], ssq[256];
  int row = blockIdx.x;
  const float* xr = x + (size_t)row * Dd;
  float sum = 0.f, sq = 0.f;
  for (int i = threadIdx.x; i < Dd; i += 256) { float v = xr[i]; sum += v; sq += v * v; }
  ssum[threadIdx.x] = sum; ssq[threadIdx.x] = sq; __syncthreads();
  for (int s = 128; s > 0; s >>= 1) {
    if (threadIdx.x < s) { ssum[threadIdx.x] += ssum[threadIdx.x + s];
                           ssq[threadIdx.x]  += ssq[threadIdx.x + s]; }
    __syncthreads();
  }
  float mean = ssum[0] * (1.f / Dd);
  float var  = ssq[0]  * (1.f / Dd) - mean * mean;
  float inv  = rsqrtf(var + 1e-5f);
  for (int i = threadIdx.x; i < Dd; i += 256)
    out[(size_t)row * Dd + i] = (xr[i] - mean) * inv * w[i] + b[i];
}

// qkv[M,2304] (col = h*192 + dh*3 + {q,k,v}) -> padded f16 Q/K [bh,208,64], V [bh,224,64]
__global__ void repack_qkv_kernel(const float* __restrict__ qkv, _Float16* __restrict__ Q,
                                  _Float16* __restrict__ Kb, _Float16* __restrict__ V) {
  int idx = blockIdx.x * 256 + threadIdx.x;
  if (idx >= BH * KPAD * DH) return;
  int dh = idx % DH, t = idx / DH;
  int bh = t / KPAD, n = t % KPAD;
  int b = bh / Hh, h = bh % Hh;
  float q = 0.f, k = 0.f, v = 0.f;
  if (n < Nt) {
    const float* src = qkv + ((size_t)b * Nt + n) * (3 * Dd) + h * (DH * 3) + dh * 3;
    q = src[0]; k = src[1]; v = src[2];
  }
  if (n < NPAD) {
    Q [((size_t)bh * NPAD + n) * DH + dh] = (_Float16)q;
    Kb[((size_t)bh * NPAD + n) * DH + dh] = (_Float16)k;
  }
  V[((size_t)bh * KPAD + n) * DH + dh] = (_Float16)v;
}

// ------------------------- attention WMMA kernels --------------------------
// S[bh,208,224] = (Q @ K^T) * 1/8 ; one wave per 16x16 tile, K-loop = 64
__global__ __launch_bounds__(128)
void attn_scores_kernel(const _Float16* __restrict__ Q, const _Float16* __restrict__ Kb,
                        float* __restrict__ S) {
  int tile = blockIdx.x * 4 + (threadIdx.x >> 5);
  if (tile >= BH * 13 * 13) return;
  int bh = tile / 169, r = tile % 169;
  int tm = r / 13, tn = r % 13;
  const _Float16* Qb = Q  + ((size_t)bh * NPAD + tm * 16) * DH;
  const _Float16* Kt = Kb + ((size_t)bh * NPAD + tn * 16) * DH;
  v8f acc = {};
#pragma unroll
  for (int k0 = 0; k0 < DH; k0 += 32) {
    v16h a = load_a_frag(Qb + k0, DH);
    v16h b = load_b_frag_wt(Kt + k0, DH);
    acc = WMMA_F16(a, b, acc);
  }
  int lane = threadIdx.x & 31;
  int nl = lane & 15, g = lane >> 4;
  float* out = S + ((size_t)bh * NPAD + tm * 16) * KPAD + tn * 16 + nl;
#pragma unroll
  for (int v = 0; v < 8; ++v)
    out[(size_t)(g * 8 + v) * KPAD] = acc[v] * 0.125f;   // 64^-0.5
}

// row softmax over 197 cols; writes f16 probs [bh,208,224], zero-pads cols/rows
__global__ __launch_bounds__(256)
void softmax_kernel(const float* __restrict__ S, _Float16* __restrict__ Pr) {
  __shared__ float red[256];
  int row = blockIdx.x;            // bh*208 + n
  int n = row % NPAD;
  const float* sr = S + (size_t)row * KPAD;
  _Float16* pr = Pr + (size_t)row * KPAD;
  if (n >= Nt) {
    for (int i = threadIdx.x; i < KPAD; i += 256) pr[i] = (_Float16)0.f;
    return;
  }
  float m = -1e30f;
  for (int i = threadIdx.x; i < Nt; i += 256) m = fmaxf(m, sr[i]);
  red[threadIdx.x] = m; __syncthreads();
  for (int s = 128; s > 0; s >>= 1) {
    if (threadIdx.x < s) red[threadIdx.x] = fmaxf(red[threadIdx.x], red[threadIdx.x + s]);
    __syncthreads();
  }
  m = red[0]; __syncthreads();
  float sum = 0.f;
  for (int i = threadIdx.x; i < Nt; i += 256) sum += expf(sr[i] - m);
  red[threadIdx.x] = sum; __syncthreads();
  for (int s = 128; s > 0; s >>= 1) {
    if (threadIdx.x < s) red[threadIdx.x] += red[threadIdx.x + s];
    __syncthreads();
  }
  float inv = 1.f / red[0];
  for (int i = threadIdx.x; i < KPAD; i += 256)
    pr[i] = (_Float16)((i < Nt) ? expf(sr[i] - m) * inv : 0.f);
}

// O[(b*197+n)*768 + h*64+dh] = P[bh,208,224] @ V[bh,224,64] ; wave per 16x16 tile
__global__ __launch_bounds__(128)
void attn_av_kernel(const _Float16* __restrict__ Pr, const _Float16* __restrict__ V,
                    float* __restrict__ O) {
  int tile = blockIdx.x * 4 + (threadIdx.x >> 5);
  if (tile >= BH * 13 * 4) return;
  int bh = tile / 52, r = tile % 52;
  int tm = r / 4, tn = r % 4;
  const _Float16* Pb = Pr + ((size_t)bh * NPAD + tm * 16) * KPAD;
  const _Float16* Vb = V  + (size_t)bh * KPAD * DH + tn * 16;
  v8f acc = {};
#pragma unroll
  for (int k0 = 0; k0 < KPAD; k0 += 32) {
    v16h a = load_a_frag(Pb + k0, KPAD);
    v16h b = load_b_frag_kn(Vb + (size_t)k0 * DH, DH);
    acc = WMMA_F16(a, b, acc);
  }
  int lane = threadIdx.x & 31;
  int nl = lane & 15, g = lane >> 4;
  int b = bh / Hh, h = bh % Hh;
  int dh = tn * 16 + nl;
#pragma unroll
  for (int v = 0; v < 8; ++v) {
    int n = tm * 16 + g * 8 + v;
    if (n < Nt) O[((size_t)b * Nt + n) * Dd + h * DH + dh] = acc[v];
  }
}

__global__ void extract_cls_kernel(const float* __restrict__ x, float* __restrict__ latent) {
  int idx = blockIdx.x * 256 + threadIdx.x;
  if (idx >= Bb * Dd) return;
  int b = idx / Dd, d = idx % Dd;
  latent[idx] = x[((size_t)b * Nt) * Dd + d];
}

// ------------------------------- launcher ----------------------------------
extern "C" void kernel_launch(void* const* d_in, const int* in_sizes, int n_in,
                              void* d_out, int out_size, void* d_ws, size_t ws_size,
                              hipStream_t stream) {
  const float* img     = (const float*)d_in[0];
  const float* patch_w = (const float*)d_in[1];
  const float* patch_b = (const float*)d_in[2];
  const float* cls_tok = (const float*)d_in[3];
  const float* pos_emb = (const float*)d_in[4];
  const float* ln1_w   = (const float*)d_in[5];
  const float* ln1_b   = (const float*)d_in[6];
  const float* qkv_w   = (const float*)d_in[7];
  const float* qkv_b   = (const float*)d_in[8];
  const float* proj_w  = (const float*)d_in[9];
  const float* proj_b  = (const float*)d_in[10];
  const float* ln2_w   = (const float*)d_in[11];
  const float* ln2_b   = (const float*)d_in[12];
  const float* ff1_w   = (const float*)d_in[13];
  const float* ff1_b   = (const float*)d_in[14];
  const float* ff2_w   = (const float*)d_in[15];
  const float* ff2_b   = (const float*)d_in[16];
  const float* head1_w = (const float*)d_in[17];
  const float* head1_b = (const float*)d_in[18];
  const float* head2_w = (const float*)d_in[19];
  const float* head2_b = (const float*)d_in[20];

  char* base = (char*)d_ws;
  size_t off = 0;
  auto alloc = [&](size_t bytes) -> char* {
    char* p = base + off;
    off += (bytes + 255) & ~(size_t)255;
    return p;
  };
  float*     x       = (float*)    alloc((size_t)Mtok * Dd * 4);
  float*     hbuf    = (float*)    alloc((size_t)Mtok * Dd * 4);
  float*     qkv     = (float*)    alloc((size_t)Mtok * 3 * Dd * 4);   // also im2col X
  _Float16*  Qf      = (_Float16*) alloc((size_t)BH * NPAD * DH * 2);
  _Float16*  Kf      = (_Float16*) alloc((size_t)BH * NPAD * DH * 2);
  _Float16*  Vf      = (_Float16*) alloc((size_t)BH * KPAD * DH * 2);
  float*     scores  = (float*)    alloc((size_t)BH * NPAD * KPAD * 4); // reused: latent/hidden
  _Float16*  probs   = (_Float16*) alloc((size_t)BH * NPAD * KPAD * 2);
  float*     attnbuf = (float*)    alloc((size_t)Mtok * Dd * 4);
  float*     ffbuf   = (float*)    alloc((size_t)Mtok * EXPD * 4);     // also patch-gemm out
  (void)ws_size; (void)in_sizes; (void)n_in; (void)out_size;

  float* X2     = qkv;                 // im2col output (19.3MB <= 58MB)
  float* Pout   = ffbuf;               // patch GEMM output (19.3MB <= 77MB)
  float* latent = scores;              // post-loop reuse
  float* hidden = scores + Bb * Dd;

  const int ACT_NONE = 0, ACT_GELU = 1, ACT_TANH = 2;
  dim3 blk256(256), blk128(128);

  auto gx = [](int M) { return (M + 127) / 128; };

  // --- patch embedding ---
  im2col_kernel<<<(Mpat * Dd + 255) / 256, blk256, 0, stream>>>(img, X2);
  gemm_bias_kernel<<<dim3(gx(Mpat), gx(Dd)), blk256, 0, stream>>>(
      X2, patch_w, patch_b, nullptr, Pout, Mpat, Dd, Dd, ACT_NONE);
  assemble_kernel<<<(Mtok * Dd + 255) / 256, blk256, 0, stream>>>(Pout, cls_tok, pos_emb, x);

  for (int l = 0; l < Ll; ++l) {
    const float* l1w = ln1_w + (size_t)l * Dd;
    const float* l1b = ln1_b + (size_t)l * Dd;
    const float* qw  = qkv_w + (size_t)l * 3 * Dd * Dd;
    const float* qb  = qkv_b + (size_t)l * 3 * Dd;
    const float* pw  = proj_w + (size_t)l * Dd * Dd;
    const float* pb  = proj_b + (size_t)l * Dd;
    const float* l2w = ln2_w + (size_t)l * Dd;
    const float* l2b = ln2_b + (size_t)l * Dd;
    const float* f1w = ff1_w + (size_t)l * EXPD * Dd;
    const float* f1b = ff1_b + (size_t)l * EXPD;
    const float* f2w = ff2_w + (size_t)l * Dd * EXPD;
    const float* f2b = ff2_b + (size_t)l * Dd;

    layernorm_kernel<<<Mtok, blk256, 0, stream>>>(x, l1w, l1b, hbuf);
    gemm_bias_kernel<<<dim3(gx(Mtok), gx(3 * Dd)), blk256, 0, stream>>>(
        hbuf, qw, qb, nullptr, qkv, Mtok, 3 * Dd, Dd, ACT_NONE);
    repack_qkv_kernel<<<(BH * KPAD * DH + 255) / 256, blk256, 0, stream>>>(qkv, Qf, Kf, Vf);
    attn_scores_kernel<<<(BH * 169 + 3) / 4, blk128, 0, stream>>>(Qf, Kf, scores);
    softmax_kernel<<<BH * NPAD, blk256, 0, stream>>>(scores, probs);
    attn_av_kernel<<<(BH * 52 + 3) / 4, blk128, 0, stream>>>(probs, Vf, attnbuf);
    gemm_bias_kernel<<<dim3(gx(Mtok), gx(Dd)), blk256, 0, stream>>>(
        attnbuf, pw, pb, x, x, Mtok, Dd, Dd, ACT_NONE);        // x += attn@pw^T + pb
    layernorm_kernel<<<Mtok, blk256, 0, stream>>>(x, l2w, l2b, hbuf);
    gemm_bias_kernel<<<dim3(gx(Mtok), gx(EXPD)), blk256, 0, stream>>>(
        hbuf, f1w, f1b, nullptr, ffbuf, Mtok, EXPD, Dd, ACT_GELU);
    gemm_bias_kernel<<<dim3(gx(Mtok), gx(Dd)), blk256, 0, stream>>>(
        ffbuf, f2w, f2b, x, x, Mtok, Dd, EXPD, ACT_NONE);      // x += gelu@f2w^T + f2b
  }

  // --- classifier head ---
  extract_cls_kernel<<<(Bb * Dd + 255) / 256, blk256, 0, stream>>>(x, latent);
  gemm_bias_kernel<<<dim3(1, gx(EXPD)), blk256, 0, stream>>>(
      latent, head1_w, head1_b, nullptr, hidden, Bb, EXPD, Dd, ACT_TANH);
  gemm_bias_kernel<<<dim3(1, gx(NCLS)), blk256, 0, stream>>>(
      hidden, head2_w, head2_b, nullptr, (float*)d_out, Bb, NCLS, EXPD, ACT_NONE);
}
